// MarioNet_6451040879201
// MI455X (gfx1250) — matrix operand order repaired and checked by reading, not compiled, not run on hardware
//
#include <hip/hip_runtime.h>

// ---------------------------------------------------------------------------
// MoE forward for MI455X (gfx1250): all large GEMMs on v_wmma_f32_16x16x32_bf16
// Compute-bound (~0.93 TFLOP vs ~0.5 GB traffic -> AI ~1800 FLOP/B).
// v4: tile staging via GLOBAL_LOAD_ASYNC_TO_LDS_B128 (ASYNCcnt) -> zero-VGPR
//     prefetch, no scratch spills. Builtin takes int4 AS1/AS3 pointers (from
//     round-3 diagnostics). Fallback keeps register prefetch with a relaxed
//     VGPR budget (__launch_bounds__(256,2)).
// ---------------------------------------------------------------------------

#define BN 4096
#define DD 1024
#define HH 2048
#define OO 1024
#define EE 12

#if defined(__has_builtin)
#if __has_builtin(__builtin_amdgcn_global_load_async_to_lds_b128) && \
    __has_builtin(__builtin_amdgcn_s_wait_asynccnt)
#define USE_ASYNC_LDS 1
#endif
#endif
#ifndef USE_ASYNC_LDS
#define USE_ASYNC_LDS 0
#endif

typedef __attribute__((ext_vector_type(4)))  int    i4;
typedef __attribute__((ext_vector_type(16))) __bf16 v16bf;
typedef __attribute__((ext_vector_type(8)))  float  v8f;

#if USE_ASYNC_LDS
typedef __attribute__((address_space(1))) i4 as1_i4;   // global int4
typedef __attribute__((address_space(3))) i4 as3_i4;   // LDS int4
#endif

union FragAB { v16bf v; unsigned int u[8]; };

__device__ __forceinline__ unsigned short f2bf(float f) {
    unsigned int x = __float_as_uint(f);
    unsigned int r = x + 0x7FFFu + ((x >> 16) & 1u);   // round-to-nearest-even
    return (unsigned short)(r >> 16);
}

// ---------------------------------------------------------------------------
// Transpose + convert: W[K,N] f32 (row-major) -> Wt[N,K] bf16 (row-major).
// 32x32 LDS tile, coalesced on both sides. K multiple of 32; N guarded.
// ---------------------------------------------------------------------------
__global__ __launch_bounds__(256)
void w_transpose_bf16_kernel(const float* __restrict__ W,
                             unsigned short* __restrict__ Wt, int Kd, int N)
{
    __shared__ float t[32][33];
    const int tx = threadIdx.x & 31;
    const int ty = threadIdx.x >> 5;          // 0..7
    const int kb = blockIdx.x * 32;
    const int nb = blockIdx.y * 32;
    #pragma unroll
    for (int j = 0; j < 4; ++j) {
        int k = kb + ty + j * 8;
        int n = nb + tx;
        t[ty + j * 8][tx] = (n < N) ? W[(size_t)k * N + n] : 0.f;
    }
    __syncthreads();
    #pragma unroll
    for (int j = 0; j < 4; ++j) {
        int n = nb + ty + j * 8;
        int k = kb + tx;
        if (n < N) Wt[(size_t)n * Kd + k] = f2bf(t[tx][ty + j * 8]);
    }
}

// ---------------------------------------------------------------------------
// Tiled GEMM: C[M,N] = act(A_bf16[M,K] @ Wt_bf16[N,K]^T + bias), epilogues.
// Block: 256 threads (8 wave32), 128x128 tile; wave tile 32x64 = 2x4 WMMA.
// Double-buffered LDS pipeline; staging is async global->LDS when available.
// EPI: 0 = f32 out, 1 = relu f32, 2 = f32 + residual (may alias C),
//      3 = bf16 out, 4 = final += w[row,e] * val  (gated MoE accumulate)
// ---------------------------------------------------------------------------
template <int EPI>
__global__ __launch_bounds__(256, 2)
void gemm_bf16_kernel(const unsigned short* __restrict__ A,
                      const unsigned short* __restrict__ Bt,
                      const float* __restrict__ bias,
                      float* Cf, unsigned short* __restrict__ Cbf,
                      const float* Res, const float* __restrict__ gw,
                      int M, int N, int Kd, int eidx)
{
    __shared__ unsigned short lsA[2][128 * 40];   // 80B pitch: 16B-aligned rows
    __shared__ unsigned short lsB[2][128 * 40];   // [n][k] (weights pre-transposed)

    const int tid  = threadIdx.x;
    const int lane = tid & 31;
    const int wid  = tid >> 5;
    const int wm   = wid & 3;     // wave row  (4 x 32 = 128 M)
    const int wn   = wid >> 2;    // wave col  (2 x 64 = 128 N)
    const int lm   = lane & 15;
    const int hi   = lane >> 4;
    const int m0   = blockIdx.y * 128;
    const int n0   = blockIdx.x * 128;

    // staging coords for this thread (2 x b128 = 16 bf16 per tile per matrix)
    int srow[2], scol[2];
    #pragma unroll
    for (int it = 0; it < 2; ++it) {
        int e8 = (it * 256 + tid) * 8;
        srow[it] = e8 >> 5;       // 0..127
        scol[it] = e8 & 31;       // {0,8,16,24}
    }

    v8f acc[2][4];
    {
        v8f z = {0.f,0.f,0.f,0.f,0.f,0.f,0.f,0.f};
        for (int i = 0; i < 2; ++i)
            for (int j = 0; j < 4; ++j) acc[i][j] = z;
    }

    const uint4 z4 = make_uint4(0u, 0u, 0u, 0u);

#if USE_ASYNC_LDS
    // stage one K-slice: global -> LDS directly, no VGPR data (ASYNCcnt)
    auto stage_async = [&](int k0, int buf) {
        #pragma unroll
        for (int it = 0; it < 2; ++it) {
            const unsigned short* ga = A + (size_t)(m0 + srow[it]) * Kd + k0 + scol[it];
            __builtin_amdgcn_global_load_async_to_lds_b128(
                (as1_i4*)ga, (as3_i4*)&lsA[buf][srow[it] * 40 + scol[it]], 0, 0);
            int gn = n0 + srow[it];
            if (gn < N) {
                const unsigned short* gb = Bt + (size_t)gn * Kd + k0 + scol[it];
                __builtin_amdgcn_global_load_async_to_lds_b128(
                    (as1_i4*)gb, (as3_i4*)&lsB[buf][srow[it] * 40 + scol[it]], 0, 0);
            } else {
                *reinterpret_cast<uint4*>(&lsB[buf][srow[it] * 40 + scol[it]]) = z4;
            }
        }
    };
    stage_async(0, 0);
    __builtin_amdgcn_s_wait_asynccnt(0);
#else
    uint4 ra[2], rb[2];
    #pragma unroll
    for (int it = 0; it < 2; ++it) {
        ra[it] = *reinterpret_cast<const uint4*>(
            A + (size_t)(m0 + srow[it]) * Kd + scol[it]);
        int gn = n0 + srow[it];
        rb[it] = (gn < N) ? *reinterpret_cast<const uint4*>(
                                Bt + (size_t)gn * Kd + scol[it])
                          : z4;
    }
    #pragma unroll
    for (int it = 0; it < 2; ++it) {
        *reinterpret_cast<uint4*>(&lsA[0][srow[it] * 40 + scol[it]]) = ra[it];
        *reinterpret_cast<uint4*>(&lsB[0][srow[it] * 40 + scol[it]]) = rb[it];
    }
#endif

    const int nk = Kd >> 5;
    for (int ki = 0; ki < nk; ++ki) {
        __syncthreads();                       // LDS[cur] ready for all waves
        const int cur = ki & 1;
        const bool more = (ki + 1) < nk;

        if (more) {                            // prefetch next slice
#if USE_ASYNC_LDS
            stage_async((ki + 1) << 5, cur ^ 1);
#else
            int k0 = (ki + 1) << 5;
            #pragma unroll
            for (int it = 0; it < 2; ++it) {
                ra[it] = *reinterpret_cast<const uint4*>(
                    A + (size_t)(m0 + srow[it]) * Kd + k0 + scol[it]);
                int gn = n0 + srow[it];
                rb[it] = (gn < N) ? *reinterpret_cast<const uint4*>(
                                        Bt + (size_t)gn * Kd + k0 + scol[it])
                                  : z4;
            }
#endif
        }

        // ---- fragments (A 16x32, B 32x16, bf16; contiguous k = 16*hi + s in both A and B) ----
        FragAB fa[2];
        #pragma unroll
        for (int sm = 0; sm < 2; ++sm) {
            int row = wm * 32 + sm * 16 + lm;
            #pragma unroll
            for (int g = 0; g < 8; ++g) {
                int kk = hi * 16 + 2 * g;
                fa[sm].u[g] = *reinterpret_cast<const unsigned int*>(
                    &lsA[cur][row * 40 + kk]);
            }
        }
        FragAB fb[4];
        #pragma unroll
        for (int sn = 0; sn < 4; ++sn) {
            int nn = wn * 64 + sn * 16 + lm;
            #pragma unroll
            for (int g = 0; g < 8; ++g) {
                int kk = hi * 16 + 2 * g;
                fb[sn].u[g] = *reinterpret_cast<const unsigned int*>(
                    &lsB[cur][nn * 40 + kk]);
            }
        }
        #pragma unroll
        for (int sm = 0; sm < 2; ++sm)
            #pragma unroll
            for (int sn = 0; sn < 4; ++sn)
                acc[sm][sn] = __builtin_amdgcn_wmma_f32_16x16x32_bf16(
                    false, fa[sm].v, false, fb[sn].v,
                    (short)0, acc[sm][sn], false, false);

        if (more) {
#if USE_ASYNC_LDS
            __builtin_amdgcn_s_wait_asynccnt(0);   // drain before barrier
#else
            const int nxt = cur ^ 1;
            #pragma unroll
            for (int it = 0; it < 2; ++it) {
                *reinterpret_cast<uint4*>(&lsA[nxt][srow[it] * 40 + scol[it]]) = ra[it];
                *reinterpret_cast<uint4*>(&lsB[nxt][srow[it] * 40 + scol[it]]) = rb[it];
            }
#endif
        }
    }

    // ---- epilogue (C layout: lane -> N, vgpr r + hi*8 -> M) ----
    #pragma unroll
    for (int sm = 0; sm < 2; ++sm) {
        #pragma unroll
        for (int sn = 0; sn < 4; ++sn) {
            int gcol = n0 + wn * 64 + sn * 16 + lm;
            if (gcol >= N) continue;
            float bv = bias[gcol];
            #pragma unroll
            for (int r = 0; r < 8; ++r) {
                int grow = m0 + wm * 32 + sm * 16 + hi * 8 + r;
                float v  = acc[sm][sn][r] + bv;
                size_t ci = (size_t)grow * N + gcol;
                if constexpr (EPI == 0) {
                    Cf[ci] = v;
                } else if constexpr (EPI == 1) {
                    Cf[ci] = fmaxf(v, 0.f);
                } else if constexpr (EPI == 2) {
                    Cf[ci] = v + Res[ci];
                } else if constexpr (EPI == 3) {
                    Cbf[ci] = f2bf(v);
                } else {
                    float wv = gw[(size_t)grow * EE + eidx];
                    Cf[ci] += wv * v;
                }
            }
        }
    }
}

// ---------------------------------------------------------------------------
// Row LayerNorm + activation -> bf16.  ACT: 0 = relu, 1 = silu
// ---------------------------------------------------------------------------
template <int ACT>
__global__ __launch_bounds__(256)
void ln_act_kernel(const float* __restrict__ X, const float* __restrict__ g,
                   const float* __restrict__ b, unsigned short* __restrict__ Y, int C)
{
    __shared__ float s1[256], s2[256];
    int row = blockIdx.x;
    const float* xr = X + (size_t)row * C;
    float s = 0.f, sq = 0.f;
    for (int c = threadIdx.x; c < C; c += 256) { float v = xr[c]; s += v; sq += v * v; }
    s1[threadIdx.x] = s; s2[threadIdx.x] = sq;
    __syncthreads();
    for (int st = 128; st > 0; st >>= 1) {
        if (threadIdx.x < st) { s1[threadIdx.x] += s1[threadIdx.x + st];
                                s2[threadIdx.x] += s2[threadIdx.x + st]; }
        __syncthreads();
    }
    float mean = s1[0] / C;
    float var  = s2[0] / C - mean * mean;
    float rstd = rsqrtf(var + 1e-5f);
    for (int c = threadIdx.x; c < C; c += 256) {
        float v = (xr[c] - mean) * rstd * g[c] + b[c];
        if constexpr (ACT == 0) v = fmaxf(v, 0.f);
        else                    v = v / (1.f + __expf(-v));
        Y[(size_t)row * C + c] = f2bf(v);
    }
}

__global__ void f32_to_bf16_kernel(const float* __restrict__ X,
                                   unsigned short* __restrict__ Y, int n)
{
    int i = blockIdx.x * blockDim.x + threadIdx.x;
    if (i < n) Y[i] = f2bf(X[i]);
}

// per-row softmax(logits / clamp(temp)) -> gate_probs, top-4 renorm -> w[B,E]
__global__ void gate_softmax_topk(const float* __restrict__ logits,
                                  const float* __restrict__ temp,
                                  float* __restrict__ gp, float* __restrict__ w)
{
    int bi = blockIdx.x * blockDim.x + threadIdx.x;
    if (bi >= BN) return;
    float t = fminf(fmaxf(temp[0], 0.5f), 5.f);
    float l[EE]; float mx = -1e30f;
    for (int e = 0; e < EE; ++e) { l[e] = logits[bi * EE + e] / t; mx = fmaxf(mx, l[e]); }
    float sum = 0.f;
    for (int e = 0; e < EE; ++e) { l[e] = __expf(l[e] - mx); sum += l[e]; }
    float p[EE];
    for (int e = 0; e < EE; ++e) { p[e] = l[e] / sum; gp[bi * EE + e] = p[e]; }
    float tmp[EE]; for (int e = 0; e < EE; ++e) tmp[e] = p[e];
    int   sel[4]; float sv[4]; float s4 = 0.f;
    for (int k = 0; k < 4; ++k) {
        int ib = 0; float bv = tmp[0];
        for (int e = 1; e < EE; ++e) if (tmp[e] > bv) { bv = tmp[e]; ib = e; }
        sel[k] = ib; sv[k] = bv; s4 += bv; tmp[ib] = -1.f;
    }
    for (int e = 0; e < EE; ++e) w[bi * EE + e] = 0.f;
    for (int k = 0; k < 4; ++k) w[bi * EE + sel[k]] = sv[k] / s4;
}

__global__ void expert_mean_kernel(const float* __restrict__ gp, float* __restrict__ u)
{
    __shared__ float sd[256];
    int e = blockIdx.x;
    float s = 0.f;
    for (int b = threadIdx.x; b < BN; b += 256) s += gp[(size_t)b * EE + e];
    sd[threadIdx.x] = s; __syncthreads();
    for (int st = 128; st > 0; st >>= 1) {
        if (threadIdx.x < st) sd[threadIdx.x] += sd[threadIdx.x + st];
        __syncthreads();
    }
    if (threadIdx.x == 0) u[e] = sd[0] / (float)BN;
}

__global__ void lb_loss_kernel(const float* __restrict__ u, float* __restrict__ out)
{
    const float ideal = 1.f / EE;
    float kl = 0.f, var = 0.f, ent = 0.f, umax = -1e30f, umin = 1e30f, usum = 0.f;
    for (int e = 0; e < EE; ++e) {
        float ue = u[e];
        kl  += ideal * (logf(ideal) - logf(ue + 1e-8f));
        var += (ue - ideal) * (ue - ideal);
        ent -= ue * logf(ue + 1e-8f);
        umax = fmaxf(umax, ue); umin = fminf(umin, ue); usum += ue;
    }
    var /= EE;
    float ent_loss = (-logf(ideal) * EE) - ent;
    float ineq = (umax - umin) * 2.f;
    float tmp[EE]; for (int e = 0; e < EE; ++e) tmp[e] = u[e];
    float qs = 0.f;
    for (int k = 0; k < EE / 3; ++k) {
        int ib = 0; float bv = tmp[0];
        for (int e = 1; e < EE; ++e) if (tmp[e] > bv) { bv = tmp[e]; ib = e; }
        qs += bv; tmp[ib] = -1e30f;
    }
    float quant  = (qs / (float)(EE / 3)) * 2.f;
    float mean_u = usum / EE;
    float sstd = 0.f;
    for (int e = 0; e < EE; ++e) { float d = u[e] - mean_u; sstd += d * d; }
    sstd = sqrtf(sstd / (EE - 1));
    float losses[5] = {kl, var, ent_loss, ineq, quant};
    float wA[5] = {0.3f, 0.25f, 0.1f, 0.25f, 0.1f};
    float wB[5] = {0.25f, 0.2f, 0.2f, 0.2f, 0.15f};
    float wC[5] = {0.2f, 0.15f, 0.3f, 0.15f, 0.2f};
    const float* ws = (umax > 0.3f) ? wA : ((sstd > 0.05f) ? wB : wC);
    float af = (umax > 0.4f) ? 2.f : ((umax > 0.25f) ? 1.5f : 1.f);
    float dot = 0.f;
    for (int i = 0; i < 5; ++i) dot += ws[i] * losses[i];
    out[0] = 0.05f * dot * af;
}

// ---------------------------------------------------------------------------
extern "C" void kernel_launch(void* const* d_in, const int* in_sizes, int n_in,
                              void* d_out, int out_size, void* d_ws, size_t ws_size,
                              hipStream_t stream)
{
    const float* x      = (const float*)d_in[0];
    const float* We_in  = (const float*)d_in[1];
    const float* be_in  = (const float*)d_in[2];
    const float* g1     = (const float*)d_in[3];
    const float* b1     = (const float*)d_in[4];
    const float* We_h1  = (const float*)d_in[5];
    const float* be_h1  = (const float*)d_in[6];
    const float* g2     = (const float*)d_in[7];
    const float* b2     = (const float*)d_in[8];
    const float* We_h2  = (const float*)d_in[9];
    const float* be_h2  = (const float*)d_in[10];
    const float* We_out = (const float*)d_in[11];
    const float* be_out = (const float*)d_in[12];
    const float* Wg0    = (const float*)d_in[13];
    const float* bg0    = (const float*)d_in[14];
    const float* gg1    = (const float*)d_in[15];
    const float* bb1    = (const float*)d_in[16];
    const float* Wg1    = (const float*)d_in[17];
    const float* bg1b   = (const float*)d_in[18];
    const float* gg2    = (const float*)d_in[19];
    const float* bb2    = (const float*)d_in[20];
    const float* Wg2    = (const float*)d_in[21];
    const float* bg2b   = (const float*)d_in[22];
    const float* Wgo    = (const float*)d_in[23];
    const float* bgo    = (const float*)d_in[24];
    const float* temp   = (const float*)d_in[25];
    (void)in_sizes; (void)n_in; (void)out_size; (void)ws_size;

    float* out_final = (float*)d_out;                     // [B,O]
    float* out_loss  = out_final + (size_t)BN * OO;       // [1]
    float* out_gp    = out_loss + 1;                      // [B,E]

    char* ws = (char*)d_ws;
    size_t off = 0;
    auto take = [&](size_t bytes) -> char* {
        char* p = ws + off;
        off = (off + bytes + 255) & ~(size_t)255;
        return p;
    };
    unsigned short* x_bf  = (unsigned short*)take((size_t)BN * DD * 2);       // x in bf16
    unsigned short* a_bf  = (unsigned short*)take((size_t)BN * HH * 2);       // LN outputs
    unsigned short* c_bf  = (unsigned short*)take((size_t)BN * (HH / 2) * 2); // e2 / h2 bf16
    unsigned short* wt_bf = (unsigned short*)take((size_t)HH * HH * 2);       // transposed weights
    float*          ef32  = (float*)take((size_t)BN * HH * 4);                // e0/e1 (also h0/h1)
    float*          logit = (float*)take((size_t)BN * EE * 4);
    float*          wgate = (float*)take((size_t)BN * EE * 4);
    float*          u     = (float*)take(256);

    dim3 blk(256);
    // transpose+convert weight matrix, then run the GEMM that consumes it
    auto gemm = [&](int epi, const unsigned short* A, const float* W, const float* bias,
                    float* Cf, unsigned short* Cbf, const float* Res, const float* gwp,
                    int M, int N, int Kd, int eidx) {
        dim3 tg(Kd / 32, (N + 31) / 32);
        w_transpose_bf16_kernel<<<tg, blk, 0, stream>>>(W, wt_bf, Kd, N);
        dim3 grid((N + 127) / 128, M / 128);
        switch (epi) {
        case 0: gemm_bf16_kernel<0><<<grid, blk, 0, stream>>>(A, wt_bf, bias, Cf, Cbf, Res, gwp, M, N, Kd, eidx); break;
        case 1: gemm_bf16_kernel<1><<<grid, blk, 0, stream>>>(A, wt_bf, bias, Cf, Cbf, Res, gwp, M, N, Kd, eidx); break;
        case 2: gemm_bf16_kernel<2><<<grid, blk, 0, stream>>>(A, wt_bf, bias, Cf, Cbf, Res, gwp, M, N, Kd, eidx); break;
        case 3: gemm_bf16_kernel<3><<<grid, blk, 0, stream>>>(A, wt_bf, bias, Cf, Cbf, Res, gwp, M, N, Kd, eidx); break;
        default: gemm_bf16_kernel<4><<<grid, blk, 0, stream>>>(A, wt_bf, bias, Cf, Cbf, Res, gwp, M, N, Kd, eidx); break;
        }
    };

    // zero the gated-accumulate output once per launch
    (void)hipMemsetAsync(out_final, 0, (size_t)BN * OO * sizeof(float), stream);

    // x -> bf16 once (reused by gating GEMM0 and every expert GEMM1)
    f32_to_bf16_kernel<<<(BN * DD + 255) / 256, blk, 0, stream>>>(x, x_bf, BN * DD);

    // ---------------- gating network ----------------
    gemm(1, x_bf, Wg0, bg0, ef32, nullptr, nullptr, nullptr, BN, 256, DD, 0);      // h0 = relu(x@Wg0+b)
    ln_act_kernel<0><<<BN, blk, 0, stream>>>(ef32, gg1, bb1, a_bf, 256);           // relu(ln(h0))
    gemm(2, a_bf, Wg1, bg1b, ef32, nullptr, ef32, nullptr, BN, 256, 256, 0);       // h1 = h0 + (..@Wg1+b)
    ln_act_kernel<0><<<BN, blk, 0, stream>>>(ef32, gg2, bb2, a_bf, 256);           // relu(ln(h1))
    gemm(3, a_bf, Wg2, bg2b, nullptr, c_bf, nullptr, nullptr, BN, 128, 256, 0);    // h2 (bf16)
    gemm(0, c_bf, Wgo, bgo, logit, nullptr, nullptr, nullptr, BN, EE, 128, 0);     // logits
    gate_softmax_topk<<<(BN + 255) / 256, blk, 0, stream>>>(logit, temp, out_gp, wgate);
    expert_mean_kernel<<<EE, blk, 0, stream>>>(out_gp, u);
    lb_loss_kernel<<<1, 1, 0, stream>>>(u, out_loss);

    // ---------------- experts: dense scan, gated accumulate ----------------
    for (int e = 0; e < EE; ++e) {
        gemm(1, x_bf, We_in + (size_t)e * DD * HH, be_in + (size_t)e * HH,
             ef32, nullptr, nullptr, nullptr, BN, HH, DD, 0);                      // e0 = relu(...)
        ln_act_kernel<0><<<BN, blk, 0, stream>>>(ef32, g1 + (size_t)e * HH,
                                                 b1 + (size_t)e * HH, a_bf, HH);   // relu(ln(e0))
        gemm(2, a_bf, We_h1 + (size_t)e * HH * HH, be_h1 + (size_t)e * HH,
             ef32, nullptr, ef32, nullptr, BN, HH, HH, 0);                         // e1 = e0 + ...
        ln_act_kernel<1><<<BN, blk, 0, stream>>>(ef32, g2 + (size_t)e * HH,
                                                 b2 + (size_t)e * HH, a_bf, HH);   // silu(ln(e1))
        gemm(3, a_bf, We_h2 + (size_t)e * HH * (HH / 2), be_h2 + (size_t)e * (HH / 2),
             nullptr, c_bf, nullptr, nullptr, BN, HH / 2, HH, 0);                  // e2 (bf16)
        gemm(4, c_bf, We_out + (size_t)e * (HH / 2) * OO, be_out + (size_t)e * OO,
             out_final, nullptr, nullptr, wgate, BN, OO, HH / 2, e);               // final += w*out
    }
}
